// MoireAttention_23819888623756
// MI455X (gfx1250) — matrix operand order repaired and checked by reading, not compiled, not run on hardware
//
#include <hip/hip_runtime.h>

typedef unsigned short u16;
typedef __attribute__((ext_vector_type(16))) __bf16 v16bf;
typedef __attribute__((ext_vector_type(8)))  float  v8f;
typedef __attribute__((ext_vector_type(8)))  unsigned short ushort8;
typedef __attribute__((ext_vector_type(16))) unsigned short ushort16;

#define B_  2
#define T_  2048
#define C_  1024
#define H_  16
#define D_  64
#define BT_ (B_ * T_)        // 4096

// ---------- helpers ----------

static __device__ __forceinline__ u16 f2bf(float f) {
  union { float f; unsigned u; } c; c.f = f;
  unsigned u = c.u;
  u += 0x7FFFu + ((u >> 16) & 1u);   // round-to-nearest-even
  return (u16)(u >> 16);
}

static __device__ __forceinline__ v8f zero8() {
  v8f z;
#pragma unroll
  for (int i = 0; i < 8; ++i) z[i] = 0.0f;
  return z;
}

// Build a v16bf fragment: elements 0..7 from lo, 8..15 from hi (8 bf16 each).
static __device__ __forceinline__ v16bf frag_from(const u16* lo, const u16* hi) {
  ushort8 a = *(const ushort8*)lo;
  ushort8 b = *(const ushort8*)hi;
  ushort16 u;
#pragma unroll
  for (int i = 0; i < 8; ++i) { u[i] = a[i]; u[i + 8] = b[i]; }
  return __builtin_bit_cast(v16bf, u);
}
static __device__ __forceinline__ v16bf frag16(const u16* p) { return frag_from(p, p + 8); }

static __device__ __forceinline__ v8f wmma_bf16(v16bf a, v16bf b, v8f c) {
  return __builtin_amdgcn_wmma_f32_16x16x32_bf16(false, a, false, b, (short)0, c, false, false);
}

// CDNA5 async global -> LDS copy, 16 bytes per lane (tracked by ASYNCcnt).
static __device__ __forceinline__ void async_b128(const void* gsrc, void* ldst) {
  unsigned ldsoff = (unsigned)(size_t)ldst;   // low 32 bits of generic ptr = LDS byte addr
  asm volatile("global_load_async_to_lds_b128 %0, %1, off"
               :: "v"(ldsoff), "v"(gsrc) : "memory");
}
static __device__ __forceinline__ void wait_async0() {
  asm volatile("s_wait_asynccnt 0x0" ::: "memory");
}

// ---------- f32 -> bf16 convert ----------

__global__ void cvt_f32_bf16(const float* __restrict__ src, u16* __restrict__ dst, int n) {
  int i = blockIdx.x * blockDim.x + threadIdx.x;
  if (i < n) dst[i] = f2bf(src[i]);
}

// ---------- generic bf16 GEMM: C(f32) = A(MxK) * B(KxN) ----------
// Block: 256 thr (8 waves), tile 128x128, K-step 32, double-buffered LDS.
// Wave grid 4x2 -> each wave computes 32x64 via 2x4 frags of 16x16.

__global__ __launch_bounds__(256)
void gemm_bf16_f32(const u16* __restrict__ A, const u16* __restrict__ Bm,
                   float* __restrict__ C, int M, int N, int K) {
  __shared__ __align__(16) u16 Alds[2][128 * 32];
  __shared__ __align__(16) u16 BldsT[2][128 * 32];   // [n][k]

  const int tid  = threadIdx.x;
  const int lane = tid & 31;
  const int w    = tid >> 5;
  const int wm   = w >> 1, wn = w & 1;
  const int m0   = blockIdx.y * 128;
  const int n0   = blockIdx.x * 128;
  const int lm   = lane & 15, lh = lane >> 4;

  // staging coordinates
  const int ar  = tid >> 1, acb  = (tid & 1) * 16;   // A: 128 rows x 32 cols
  const int bkr = tid >> 3, bncb = (tid & 7) * 16;   // B: 32 rows x 128 cols
  const u16* aSrc = A  + (size_t)(m0 + ar) * K + acb;
  const u16* bSrc = Bm + (size_t)bkr * N + n0 + bncb;

  v8f acc[2][4];
#pragma unroll
  for (int i = 0; i < 2; ++i)
#pragma unroll
    for (int j = 0; j < 4; ++j) acc[i][j] = zero8();

  // ---- prologue: stage tile 0 into buffer 0 ----
  async_b128(aSrc,     &Alds[0][ar * 32 + acb]);
  async_b128(aSrc + 8, &Alds[0][ar * 32 + acb + 8]);
  {
    ushort8 b0 = *(const ushort8*)bSrc;
    ushort8 b1 = *(const ushort8*)(bSrc + 8);
#pragma unroll
    for (int i = 0; i < 8; ++i) {
      BldsT[0][(bncb + i) * 32 + bkr]     = b0[i];
      BldsT[0][(bncb + 8 + i) * 32 + bkr] = b1[i];
    }
  }
  wait_async0();
  __syncthreads();

  int buf = 0;
  for (int k0 = 0; k0 < K; k0 += 32) {
    const int  nb      = buf ^ 1;
    const bool hasNext = (k0 + 32) < K;

    ushort8 nb0, nb1;
    if (hasNext) {
      // issue next A tile (async, straight to LDS) + next B tile (to VGPRs)
      const u16* an = aSrc + (k0 + 32);
      async_b128(an,     &Alds[nb][ar * 32 + acb]);
      async_b128(an + 8, &Alds[nb][ar * 32 + acb + 8]);
      const u16* bn = bSrc + (size_t)(k0 + 32) * N;
      nb0 = *(const ushort8*)bn;
      nb1 = *(const ushort8*)(bn + 8);
    }

    // ---- compute from current buffer ----
    v16bf af[2];
#pragma unroll
    for (int i = 0; i < 2; ++i) {
      const u16* p = &Alds[buf][(wm * 32 + i * 16 + lm) * 32] + lh * 8;
      af[i] = frag_from(p, p + 16);
    }
#pragma unroll
    for (int j = 0; j < 4; ++j) {
      const u16* p = &BldsT[buf][(wn * 64 + j * 16 + lm) * 32 + lh * 8];
      v16bf bf = frag_from(p, p + 16);
#pragma unroll
      for (int i = 0; i < 2; ++i) acc[i][j] = wmma_bf16(af[i], bf, acc[i][j]);
    }

    if (hasNext) {
      // transpose-scatter next B tile into the back buffer
#pragma unroll
      for (int i = 0; i < 8; ++i) {
        BldsT[nb][(bncb + i) * 32 + bkr]     = nb0[i];
        BldsT[nb][(bncb + 8 + i) * 32 + bkr] = nb1[i];
      }
      wait_async0();
    }
    __syncthreads();
    buf = nb;
  }

#pragma unroll
  for (int i = 0; i < 2; ++i)
#pragma unroll
    for (int j = 0; j < 4; ++j)
#pragma unroll
      for (int v = 0; v < 8; ++v) {
        int row = m0 + wm * 32 + i * 16 + v + 8 * lh;
        int col = n0 + wn * 64 + j * 16 + lm;
        C[(size_t)row * N + col] = acc[i][j][v];
      }
}

// ---------- transforms ----------
// raw (BT x 2C): [0,C) amp, [C,2C) phase ->  out (B,H,T,128) bf16 [real|imag], scaled

__global__ void transform_qk(const float* __restrict__ raw, u16* __restrict__ out, float scale) {
  int i = blockIdx.x * 256 + threadIdx.x;          // [0, BT*C)
  if (i >= BT_ * C_) return;
  int c  = i & (C_ - 1);
  int bt = i >> 10;
  int b  = bt >> 11, t = bt & (T_ - 1);
  int h  = c >> 6,  d = c & 63;
  float a  = raw[(size_t)bt * (2 * C_) + c];
  float ph = raw[(size_t)bt * (2 * C_) + C_ + c];
  float amp = (a > 15.0f) ? a : __logf(1.0f + __expf(a));
  amp *= scale;
  size_t base = (((size_t)(b * H_ + h)) * T_ + t) * 128 + d;
  out[base]      = f2bf(amp * __cosf(ph));
  out[base + 64] = f2bf(amp * __sinf(ph));
}

__global__ void transform_v(const float* __restrict__ raw, u16* __restrict__ out) {
  int i = blockIdx.x * 256 + threadIdx.x;          // [0, BT*C)
  if (i >= BT_ * C_) return;
  int c  = i & (C_ - 1);
  int bt = i >> 10;
  int b  = bt >> 11, t = bt & (T_ - 1);
  int h  = c >> 6,  d = c & 63;
  out[(((size_t)(b * H_ + h)) * T_ + t) * 64 + d] = f2bf(raw[(size_t)bt * C_ + c]);
}

// ---------- flash attention with theta gate ----------
// Grid: (T/128, B*H). 256 thr = 8 waves; wave w owns 16 query rows.

__global__ __launch_bounds__(256)
void flash_moire(const u16* __restrict__ Qc, const u16* __restrict__ Kc,
                 const u16* __restrict__ Vh, const float* __restrict__ theta_off,
                 u16* __restrict__ Obf) {
  __shared__ __align__(16) u16 Klds[64 * 128];     // [s][c]
  __shared__ __align__(16) u16 Vtlds[64 * 64];     // [d][s]
  __shared__ __align__(16) u16 Plds[8][16 * 64];   // per-wave P tile

  const int tid  = threadIdx.x;
  const int lane = tid & 31;
  const int w    = tid >> 5;
  const int lm   = lane & 15, lh = lane >> 4;
  const int bh   = blockIdx.y;
  const int b    = bh >> 4, h = bh & (H_ - 1);
  const int tblk = blockIdx.x * 128;
  const int trow = tblk + w * 16;
  const float theta = theta_off[h];

  // Q fragments: row strip of 16, K-dim 128 -> 4 A-frags
  v16bf qf[4];
  {
    const u16* qb = Qc + ((size_t)bh * T_ + trow + lm) * 128;
#pragma unroll
    for (int kk = 0; kk < 4; ++kk) {
      const u16* p = qb + kk * 32 + lh * 8;
      qf[kk] = frag_from(p, p + 16);
    }
  }

  v8f Ofr[4];
#pragma unroll
  for (int j = 0; j < 4; ++j) Ofr[j] = zero8();
  float mrow[8], lrow[8];
#pragma unroll
  for (int v = 0; v < 8; ++v) { mrow[v] = -1e30f; lrow[v] = 0.0f; }

  const int nkt = 2 * blockIdx.x + 2;
  for (int kt = 0; kt < nkt; ++kt) {
    const int s0 = kt * 64;
    // stage K tile (64x128) row-major via async global->LDS (64B/thread)
    {
      int sr = tid >> 2, cb = (tid & 3) * 32;
      const u16* src = Kc + ((size_t)bh * T_ + s0 + sr) * 128 + cb;
#pragma unroll
      for (int q = 0; q < 4; ++q)
        async_b128(src + q * 8, &Klds[sr * 128 + cb + q * 8]);
    }
    // stage V tile transposed (d, s) via VGPR scatter
    {
      int sr = tid >> 2, db = (tid & 3) * 16;
      const u16* src = Vh + ((size_t)bh * T_ + s0 + sr) * 64 + db;
      ushort8 v0 = *(const ushort8*)src;
      ushort8 v1 = *(const ushort8*)(src + 8);
#pragma unroll
      for (int i = 0; i < 8; ++i) {
        Vtlds[(db + i) * 64 + sr]     = v0[i];
        Vtlds[(db + 8 + i) * 64 + sr] = v1[i];
      }
    }
    wait_async0();
    __syncthreads();

    if (s0 <= trow + 15) {
      // S = Q * K^T   (16x64 per wave)
      v8f S[4];
#pragma unroll
      for (int j = 0; j < 4; ++j) S[j] = zero8();
#pragma unroll
      for (int j = 0; j < 4; ++j)
#pragma unroll
        for (int kk = 0; kk < 4; ++kk) {
          const u16* p = &Klds[(j * 16 + lm) * 128 + kk * 32 + lh * 8];
          S[j] = wmma_bf16(qf[kk], frag_from(p, p + 16), S[j]);
        }

      // theta gate + causal mask
#pragma unroll
      for (int j = 0; j < 4; ++j)
#pragma unroll
        for (int v = 0; v < 8; ++v) {
          int t = trow + v + 8 * lh;
          int s = s0 + j * 16 + lm;
          float sv = S[j][v];
          sv = (s > t) ? -1e30f
                       : sv * __cosf(theta * (float)(s - t) * 0.125f);
          S[j][v] = sv;
        }

      // online softmax: row stats via 16-lane shfl reductions
      float alpha[8];
#pragma unroll
      for (int v = 0; v < 8; ++v) {
        float r = fmaxf(fmaxf(S[0][v], S[1][v]), fmaxf(S[2][v], S[3][v]));
        r = fmaxf(r, __shfl_xor(r, 1, 32));
        r = fmaxf(r, __shfl_xor(r, 2, 32));
        r = fmaxf(r, __shfl_xor(r, 4, 32));
        r = fmaxf(r, __shfl_xor(r, 8, 32));
        float mn = fmaxf(mrow[v], r);
        alpha[v] = __expf(mrow[v] - mn);
        mrow[v]  = mn;
      }
#pragma unroll
      for (int j = 0; j < 4; ++j)
#pragma unroll
        for (int v = 0; v < 8; ++v)
          S[j][v] = __expf(S[j][v] - mrow[v]);
#pragma unroll
      for (int v = 0; v < 8; ++v) {
        float r = (S[0][v] + S[1][v]) + (S[2][v] + S[3][v]);
        r += __shfl_xor(r, 1, 32);
        r += __shfl_xor(r, 2, 32);
        r += __shfl_xor(r, 4, 32);
        r += __shfl_xor(r, 8, 32);
        lrow[v] = lrow[v] * alpha[v] + r;
      }
#pragma unroll
      for (int j = 0; j < 4; ++j)
#pragma unroll
        for (int v = 0; v < 8; ++v) Ofr[j][v] *= alpha[v];

      // P: C-frag layout -> bf16 in per-wave LDS (re-layout for A operand)
#pragma unroll
      for (int j = 0; j < 4; ++j)
#pragma unroll
        for (int v = 0; v < 8; ++v)
          Plds[w][(v + 8 * lh) * 64 + j * 16 + lm] = f2bf(S[j][v]);

      // O += P * V
#pragma unroll
      for (int j = 0; j < 4; ++j)
#pragma unroll
        for (int kk = 0; kk < 2; ++kk) {
          const u16* pa = &Plds[w][lm * 64 + kk * 32 + lh * 8];
          v16bf af = frag_from(pa, pa + 16);
          const u16* pb = &Vtlds[(j * 16 + lm) * 64 + kk * 32 + lh * 8];
          Ofr[j] = wmma_bf16(af, frag_from(pb, pb + 16), Ofr[j]);
        }
    }
    __syncthreads();
  }

  // normalize and store as (B, T, C) bf16
#pragma unroll
  for (int v = 0; v < 8; ++v) {
    float inv = 1.0f / lrow[v];
    int t = trow + v + 8 * lh;
    u16* dst = Obf + ((size_t)b * T_ + t) * C_ + h * 64;
#pragma unroll
    for (int j = 0; j < 4; ++j) dst[j * 16 + lm] = f2bf(Ofr[j][v] * inv);
  }
}

// ---------- launch ----------

extern "C" void kernel_launch(void* const* d_in, const int* in_sizes, int n_in,
                              void* d_out, int out_size, void* d_ws, size_t ws_size,
                              hipStream_t stream) {
  const float* x     = (const float*)d_in[0];
  const float* Wq    = (const float*)d_in[1];
  const float* Wk    = (const float*)d_in[2];
  const float* Wv    = (const float*)d_in[3];
  const float* Wo    = (const float*)d_in[4];
  const float* theta = (const float*)d_in[5];

  char* ws = (char*)d_ws;
  size_t off = 0;
  auto alloc = [&](size_t bytes) -> void* {
    void* p = (void*)(ws + off);
    off += (bytes + 255) & ~(size_t)255;
    return p;
  };

  u16*   xbf   = (u16*)alloc((size_t)BT_ * C_ * 2);
  u16*   wqbf  = (u16*)alloc((size_t)C_ * 2 * C_ * 2);
  u16*   wkbf  = (u16*)alloc((size_t)C_ * 2 * C_ * 2);
  u16*   wvbf  = (u16*)alloc((size_t)C_ * C_ * 2);
  u16*   wobf  = (u16*)alloc((size_t)C_ * C_ * 2);
  float* qraw  = (float*)alloc((size_t)BT_ * 2 * C_ * 4);
  float* kraw  = (float*)alloc((size_t)BT_ * 2 * C_ * 4);
  float* vraw  = (float*)alloc((size_t)BT_ * C_ * 4);
  u16*   Qc    = (u16*)alloc((size_t)B_ * H_ * T_ * 128 * 2);
  u16*   Kc    = (u16*)alloc((size_t)B_ * H_ * T_ * 128 * 2);
  u16*   Vh    = (u16*)alloc((size_t)B_ * H_ * T_ * 64 * 2);
  u16*   Obf   = (u16*)alloc((size_t)BT_ * C_ * 2);

  // 1. convert operands to bf16
  {
    int n;
    n = BT_ * C_;       cvt_f32_bf16<<<(n + 255) / 256, 256, 0, stream>>>(x,  xbf,  n);
    n = C_ * 2 * C_;    cvt_f32_bf16<<<(n + 255) / 256, 256, 0, stream>>>(Wq, wqbf, n);
    n = C_ * 2 * C_;    cvt_f32_bf16<<<(n + 255) / 256, 256, 0, stream>>>(Wk, wkbf, n);
    n = C_ * C_;        cvt_f32_bf16<<<(n + 255) / 256, 256, 0, stream>>>(Wv, wvbf, n);
    n = C_ * C_;        cvt_f32_bf16<<<(n + 255) / 256, 256, 0, stream>>>(Wo, wobf, n);
  }

  // 2. projections (WMMA GEMMs)
  gemm_bf16_f32<<<dim3((2 * C_) / 128, BT_ / 128), 256, 0, stream>>>(xbf, wqbf, qraw, BT_, 2 * C_, C_);
  gemm_bf16_f32<<<dim3((2 * C_) / 128, BT_ / 128), 256, 0, stream>>>(xbf, wkbf, kraw, BT_, 2 * C_, C_);
  gemm_bf16_f32<<<dim3(C_ / 128,       BT_ / 128), 256, 0, stream>>>(xbf, wvbf, vraw, BT_, C_, C_);

  // 3. softplus/phase transforms (SCALE = 1/sqrt(D) folded into Q)
  {
    int n = BT_ * C_;
    transform_qk<<<(n + 255) / 256, 256, 0, stream>>>(qraw, Qc, 0.125f);
    transform_qk<<<(n + 255) / 256, 256, 0, stream>>>(kraw, Kc, 1.0f);
    transform_v <<<(n + 255) / 256, 256, 0, stream>>>(vraw, Vh);
  }

  // 4. gated causal flash attention
  flash_moire<<<dim3(T_ / 128, B_ * H_), 256, 0, stream>>>(Qc, Kc, Vh, theta, Obf);

  // 5. output projection -> f32 d_out
  gemm_bf16_f32<<<dim3(C_ / 128, BT_ / 128), 256, 0, stream>>>(Obf, wobf, (float*)d_out, BT_, C_, C_);
}
